// Decoder_4372276707342
// MI455X (gfx1250) — matrix operand order repaired and checked by reading, not compiled, not run on hardware
//
#include <hip/hip_runtime.h>

namespace {

constexpr int CB = 2;     // batch
constexpr int CS = 1024;  // sequence
constexpr int CD = 1024;  // model dim
constexpr int CH = 16;    // heads
constexpr int CL = 2;     // layers
constexpr int CDF = 4096; // ffn dim
constexpr int CDK = 64;   // head dim

typedef _Float16 v16h __attribute__((ext_vector_type(16)));
typedef float v8f __attribute__((ext_vector_type(8)));
typedef _Float16 h2 __attribute__((ext_vector_type(2)));
typedef int v4i __attribute__((ext_vector_type(4)));

constexpr int BM = 64, BN = 64, BK = 32;
constexpr int LDH = BK + 8; // 40 halves = 80B row pitch: 16B-aligned rows, bank-skewed

union Frag16 { v16h v; h2 p[8]; };

#define GLOBAL_AS __attribute__((address_space(1)))
#define LDS_AS __attribute__((address_space(3)))

#if defined(__has_builtin)
#if __has_builtin(__builtin_amdgcn_global_load_async_to_lds_b128) && \
    __has_builtin(__builtin_amdgcn_s_wait_asynccnt)
#define USE_ASYNC_LDS 1
#endif
#endif

// ---------------------------------------------------------------------------
// Batched GEMM, f16 operands, fp32 accumulate via v_wmma_f32_16x16x32_f16.
//   A: M x K f16 row-major (lda).  B: ALWAYS N x K f16 row-major (ldb), i.e.
//   pre-transposed — every operand is K-contiguous so tiles stage as one
//   b128 per thread (async-to-LDS on gfx1250 when available, ASYNCcnt-waited),
//   double-buffered ping-pong LDS to overlap stage(k+1) with compute(k).
// Epilogue: x = alpha*acc + bias[n]; flags: 1=ReLU, 2=f16 out, 4=f16 out
// transposed (C[n][m], used to produce V^T). grid.z -> (b,h) via batchH with
// per-b/per-h strides for A/B/C/bias. All dims are tile multiples here.
// ---------------------------------------------------------------------------
__global__ __launch_bounds__(128) void gemm_f16_wmma(
    const _Float16* __restrict__ A, const _Float16* __restrict__ Bm,
    const float* __restrict__ bias, void* __restrict__ Cv,
    int K, int lda, int ldb, int ldc,
    long long sAb, long long sAh, long long sBb, long long sBh,
    long long sCb, long long sCh, long long sBiasH,
    int batchH, float alpha, int flags)
{
  __shared__ __align__(16) _Float16 As[2][BM][LDH];
  __shared__ __align__(16) _Float16 Bt[2][BN][LDH];

  const int z  = blockIdx.z;
  const int bb = z / batchH, hh = z % batchH;
  const _Float16* Ap = A  + bb * sAb + hh * sAh;
  const _Float16* Bp = Bm + bb * sBb + hh * sBh;
  const float* biasp = bias ? (bias + hh * sBiasH) : nullptr;

  const int m0 = blockIdx.y * BM, n0 = blockIdx.x * BN;
  const int t = threadIdx.x;
  const int wave = t >> 5, lane = t & 31;
  const int wm = (wave >> 1) << 5, wn = (wave & 1) << 5;
  const int l16 = lane & 15, hi = lane >> 4;

  // tile staging: 64 rows x 32 halves = 128 threads x one 16-byte chunk each
  const int lr = t >> 1;           // row 0..63
  const int lc = (t & 1) << 4;     // half-offset 0 or 16 within the row
  const long long aoff = (long long)(m0 + lr) * lda + lc;
  const long long boff = (long long)(n0 + lr) * ldb + lc;

  const int nk = K / BK;
  v8f acc[2][2] = {};

  auto stage = [&](int kb, int buf) {
#ifdef USE_ASYNC_LDS
    __builtin_amdgcn_global_load_async_to_lds_b128(
        (GLOBAL_AS v4i*)(void*)(Ap + aoff + (long long)kb * BK),
        (LDS_AS v4i*)&As[buf][lr][lc], 0, 0);
    __builtin_amdgcn_global_load_async_to_lds_b128(
        (GLOBAL_AS v4i*)(void*)(Bp + boff + (long long)kb * BK),
        (LDS_AS v4i*)&Bt[buf][lr][lc], 0, 0);
#else
    *(v4i*)&As[buf][lr][lc] = *(const v4i*)(Ap + aoff + (long long)kb * BK);
    *(v4i*)&Bt[buf][lr][lc] = *(const v4i*)(Bp + boff + (long long)kb * BK);
#endif
  };

  stage(0, 0);
  for (int kb = 0; kb < nk; ++kb) {
    const int cur = kb & 1;
    if (kb + 1 < nk) stage(kb + 1, cur ^ 1);
#ifdef USE_ASYNC_LDS
    if (kb + 1 < nk) __builtin_amdgcn_s_wait_asynccnt(2);  // stage(kb) done
    else             __builtin_amdgcn_s_wait_asynccnt(0);
#endif
    __syncthreads();

#pragma unroll
    for (int mi = 0; mi < 2; ++mi) {
      // A frag 16x32: lane l16 = M row; VGPR v -> k = (v/4)*16 + hi*8 + (v%4)*2
      Frag16 af;
      const int arow = wm + (mi << 4) + l16;
#pragma unroll
      for (int v = 0; v < 8; ++v) {
        const int kk = ((v >> 2) << 4) + (hi << 3) + ((v & 3) << 1);
        af.p[v] = *(const h2*)&As[cur][arow][kk];
      }
#pragma unroll
      for (int ni = 0; ni < 2; ++ni) {
        // B frag 32x16: lane l16 = N col; VGPR v -> k = (v/4)*16 + hi*8 + (v%4)*2
        Frag16 bf;
        const int bcol = wn + (ni << 4) + l16;
#pragma unroll
        for (int v = 0; v < 8; ++v) {
          const int kk = ((v >> 2) << 4) + (hi << 3) + ((v & 3) << 1);
          bf.p[v] = *(const h2*)&Bt[cur][bcol][kk];
        }
        acc[mi][ni] = __builtin_amdgcn_wmma_f32_16x16x32_f16(
            false, af.v, false, bf.v, (short)0, acc[mi][ni], false, false);
      }
    }
    __syncthreads();  // everyone done reading buf cur before it is restaged
  }

  float* Cf = (float*)Cv;
  _Float16* Ch = (_Float16*)Cv;
  const long long cb = bb * sCb + hh * sCh;
#pragma unroll
  for (int mi = 0; mi < 2; ++mi) {
#pragma unroll
    for (int ni = 0; ni < 2; ++ni) {
      const int n = n0 + wn + (ni << 4) + l16;
      const float bv = biasp ? biasp[n] : 0.0f;
#pragma unroll
      for (int r = 0; r < 8; ++r) {
        const int m = m0 + wm + (mi << 4) + (hi << 3) + r;
        float x = alpha * acc[mi][ni][r] + bv;
        if (flags & 1) x = x > 0.0f ? x : 0.0f;
        if (flags & 4)      Ch[cb + (long long)n * ldc + m] = (_Float16)x;
        else if (flags & 2) Ch[cb + (long long)m * ldc + n] = (_Float16)x;
        else                Cf[cb + (long long)m * ldc + n] = x;
      }
    }
  }
}

// ---------------------------------------------------------------------------
// K x N fp32 -> N x K f16 tiled transpose (one-time weight prep per launch).
// ---------------------------------------------------------------------------
__global__ __launch_bounds__(256) void transpose_cvt(
    const float* __restrict__ in, _Float16* __restrict__ out, int K, int N)
{
  __shared__ _Float16 tile[32][34];
  const long long base = (long long)blockIdx.z * K * N;
  const int k0 = blockIdx.y << 5, n0 = blockIdx.x << 5;
  const int tx = threadIdx.x & 31, ty = threadIdx.x >> 5;
#pragma unroll
  for (int i = 0; i < 4; ++i) {
    const int k = ty + (i << 3);
    tile[k][tx] = (_Float16)in[base + (long long)(k0 + k) * N + (n0 + tx)];
  }
  __syncthreads();
#pragma unroll
  for (int i = 0; i < 4; ++i) {
    const int n = ty + (i << 3);
    out[base + (long long)(n0 + n) * K + (k0 + tx)] = tile[tx][n];
  }
}

__global__ __launch_bounds__(256) void cvt_f16(
    const float* __restrict__ in, _Float16* __restrict__ out, long long n)
{
  long long i = (long long)blockIdx.x * blockDim.x + threadIdx.x;
  const long long stride = (long long)gridDim.x * blockDim.x;
  for (; i < n; i += stride) out[i] = (_Float16)in[i];
}

// ---------------------------------------------------------------------------
// Row softmax over S=1024 with optional causal mask; writes normalized probs
// as f16 IN PLACE over the fp32 row (halves 0..S-1 occupy the first half of
// the row; all reads complete before the reduction barriers, so safe).
// ---------------------------------------------------------------------------
__global__ __launch_bounds__(256) void softmax_causal(float* __restrict__ S,
                                                      int ncols, int causal)
{
  const int row = blockIdx.x;
  const int qpos = row & (ncols - 1);
  float* p = S + (long long)row * ncols;
  const int t = threadIdx.x;
  __shared__ float red[256];

  float vals[4];
  int dead[4];
  float vmax = -3.402823466e38f;
#pragma unroll
  for (int i = 0; i < 4; ++i) {
    const int c = t + (i << 8);
    dead[i] = causal && (c > qpos);
    const float x = dead[i] ? -3.402823466e38f : p[c];
    vals[i] = x;
    vmax = fmaxf(vmax, x);
  }
  red[t] = vmax; __syncthreads();
  for (int s = 128; s > 0; s >>= 1) {
    if (t < s) red[t] = fmaxf(red[t], red[t + s]);
    __syncthreads();
  }
  const float rowmax = red[0];
  __syncthreads();

  float sum = 0.0f;
#pragma unroll
  for (int i = 0; i < 4; ++i) {
    const float e = dead[i] ? 0.0f : __expf(vals[i] - rowmax);
    vals[i] = e; sum += e;
  }
  red[t] = sum; __syncthreads();
  for (int s = 128; s > 0; s >>= 1) {
    if (t < s) red[t] += red[t + s];
    __syncthreads();
  }
  const float inv = red[0] > 0.0f ? 1.0f / red[0] : 0.0f;
  _Float16* ph = (_Float16*)p;
#pragma unroll
  for (int i = 0; i < 4; ++i) ph[t + (i << 8)] = (_Float16)(vals[i] * inv);
}

// ---------------------------------------------------------------------------
// outf/outh = LayerNorm(c0*a + c1*b + c2*c)*g + beta; fp32 + f16 outputs.
// Row cached in registers -> in-place (outf aliasing b) is safe.
// ---------------------------------------------------------------------------
__global__ __launch_bounds__(256) void resid_layernorm(
    const float* __restrict__ a, const float* __restrict__ b,
    const float* __restrict__ c, float c0, float c1, float c2,
    const float* __restrict__ g, const float* __restrict__ beta,
    float* __restrict__ outf, _Float16* __restrict__ outh, int ncols)
{
  const int row = blockIdx.x;
  const long long off = (long long)row * ncols;
  const int t = threadIdx.x;
  __shared__ float red[256];

  float vals[4];
  float s = 0.0f;
#pragma unroll
  for (int i = 0; i < 4; ++i) {
    const int idx = t + (i << 8);
    float x = c0 * a[off + idx] + c1 * b[off + idx];
    if (c) x += c2 * c[off + idx];
    vals[i] = x; s += x;
  }
  red[t] = s; __syncthreads();
  for (int k = 128; k > 0; k >>= 1) { if (t < k) red[t] += red[t + k]; __syncthreads(); }
  const float mean = red[0] * (1.0f / ncols);
  __syncthreads();

  float vs = 0.0f;
#pragma unroll
  for (int i = 0; i < 4; ++i) { const float d = vals[i] - mean; vs += d * d; }
  red[t] = vs; __syncthreads();
  for (int k = 128; k > 0; k >>= 1) { if (t < k) red[t] += red[t + k]; __syncthreads(); }
  const float inv = rsqrtf(red[0] * (1.0f / ncols) + 1e-5f);
#pragma unroll
  for (int i = 0; i < 4; ++i) {
    const int idx = t + (i << 8);
    const float y = (vals[i] - mean) * inv * g[idx] + beta[idx];
    outf[off + idx] = y;
    outh[off + idx] = (_Float16)y;
  }
}

} // namespace

extern "C" void kernel_launch(void* const* d_in, const int* in_sizes, int n_in,
                              void* d_out, int out_size, void* d_ws, size_t ws_size,
                              hipStream_t stream)
{
  (void)in_sizes; (void)n_in; (void)out_size; (void)ws_size;
  const float* x   = (const float*)d_in[0];
  const float* enc = (const float*)d_in[1];
  const float* peo = (const float*)d_in[2];
  const float* pdo = (const float*)d_in[3];

  const long long ACT = (long long)CB * CS * CD;       // 2M elements
  const long long SC  = (long long)CB * CH * CS * CS;  // 32M elements
  const long long FFE = (long long)CB * CS * CDF;      // 8M elements

  // ---- fp32 workspace ----
  float* f = (float*)d_ws;
  float* scb  = f; f += SC;   // scores (softmax rewrites rows as f16 probs)
  float* outa = f; f += ACT;
  float* outb = f; f += ACT;
  float* hbuf = f; f += ACT;
  float* r1b  = f; f += ACT;
  // ---- f16 workspace ----
  _Float16* hp = (_Float16*)f;
  auto takeh = [&](long long n) { _Float16* p = hp; hp += n; return p; };
  _Float16* hh   = takeh(ACT);  // current hidden state, f16
  _Float16* r1h  = takeh(ACT);
  _Float16* qh   = takeh(ACT);
  _Float16* kh   = takeh(ACT);
  _Float16* vth  = takeh(ACT);  // V^T per head: [b,h,e,s]
  _Float16* ctxh = takeh(ACT);
  _Float16* cath = takeh(ACT);
  _Float16* ench = takeh(ACT);
  _Float16* peh  = takeh(ACT);
  _Float16* pdh  = takeh(ACT);
  _Float16* ffh  = takeh(FFE);

  // ---- one-time (per launch) weight transpose+convert: W[K][N] -> WT[N][K] f16
  const int attnBase[4] = {4, 14, 24, 34};
  _Float16 *wqT[CL][4], *wkT[CL][4], *wvT[CL][4], *whT[CL][4], *woT[CL][4];
  _Float16 *w1T[CL], *w2T[CL];
  for (int l = 0; l < CL; ++l) {
    for (int a = 0; a < 4; ++a) {
      const float* Wq = (const float*)d_in[attnBase[a] + 0] + (long long)l * CH * CD * CDK;
      const float* Wk = (const float*)d_in[attnBase[a] + 2] + (long long)l * CH * CD * CDK;
      const float* Wv = (const float*)d_in[attnBase[a] + 4] + (long long)l * CH * CD * CDK;
      const float* Wh = (const float*)d_in[attnBase[a] + 6] + (long long)l * CH * CDK * CDK;
      const float* Wo = (const float*)d_in[attnBase[a] + 8] + (long long)l * CH * CDK * CD;
      wqT[l][a] = takeh((long long)CH * CD * CDK);
      transpose_cvt<<<dim3(CDK / 32, CD / 32, CH), 256, 0, stream>>>(Wq, wqT[l][a], CD, CDK);
      wkT[l][a] = takeh((long long)CH * CD * CDK);
      transpose_cvt<<<dim3(CDK / 32, CD / 32, CH), 256, 0, stream>>>(Wk, wkT[l][a], CD, CDK);
      wvT[l][a] = takeh((long long)CH * CD * CDK);
      transpose_cvt<<<dim3(CDK / 32, CD / 32, CH), 256, 0, stream>>>(Wv, wvT[l][a], CD, CDK);
      whT[l][a] = takeh((long long)CH * CDK * CDK);
      transpose_cvt<<<dim3(CDK / 32, CDK / 32, CH), 256, 0, stream>>>(Wh, whT[l][a], CDK, CDK);
      woT[l][a] = takeh((long long)CD * CD);
      transpose_cvt<<<dim3(CD / 32, CD / 32, 1), 256, 0, stream>>>(Wo, woT[l][a], CD, CD);
    }
    const float* W1 = (const float*)d_in[50] + (long long)l * CD * CDF;
    const float* W2 = (const float*)d_in[52] + (long long)l * CDF * CD;
    w1T[l] = takeh((long long)CD * CDF);
    transpose_cvt<<<dim3(CDF / 32, CD / 32, 1), 256, 0, stream>>>(W1, w1T[l], CD, CDF);
    w2T[l] = takeh((long long)CDF * CD);
    transpose_cvt<<<dim3(CD / 32, CDF / 32, 1), 256, 0, stream>>>(W2, w2T[l], CDF, CD);
  }

  // ---- activation converts + fp32 copy of h ----
  cvt_f16<<<1024, 256, 0, stream>>>(x, hh, ACT);
  cvt_f16<<<1024, 256, 0, stream>>>(enc, ench, ACT);
  cvt_f16<<<1024, 256, 0, stream>>>(peo, peh, ACT);
  cvt_f16<<<1024, 256, 0, stream>>>(pdo, pdh, ACT);
  (void)hipMemcpyAsync(hbuf, x, ACT * sizeof(float), hipMemcpyDeviceToDevice, stream);

  auto gemm = [&](const _Float16* A, const _Float16* Bm, const float* bias, void* C,
                  int M, int N, int K, int lda, int ldb, int ldc,
                  long long sAb, long long sAh, long long sBb, long long sBh,
                  long long sCb, long long sCh, long long sBiasH,
                  int batches, int batchH, float alpha, int flags) {
    dim3 grid(N / BN, M / BM, batches);
    gemm_f16_wmma<<<grid, 128, 0, stream>>>(A, Bm, bias, C, K, lda, ldb, ldc,
        sAb, sAh, sBb, sBh, sCb, sCh, sBiasH, batchH, alpha, flags);
  };

  const long long sAct = (long long)CS * CD;   // per-b activation stride
  const long long sHd  = (long long)CS * CDK;  // per-(b,h) head stride
  const long long sBH  = (long long)CH * CS * CDK;

  auto mha = [&](const _Float16* xqh, const _Float16* xkvh,
                 const _Float16* WqT, const _Float16* WkT, const _Float16* WvT,
                 const _Float16* WhT, const _Float16* WoT,
                 const float* bq, const float* bk, const float* bv,
                 const float* bh_, const float* bo_, int causal, float* out) {
    gemm(xqh, WqT, bq, qh, CS, CDK, CD, CD, CD, CDK,
         sAct, 0, 0, (long long)CDK * CD, sBH, sHd, CDK, CB * CH, CH, 1.0f, 2);
    gemm(xkvh, WkT, bk, kh, CS, CDK, CD, CD, CD, CDK,
         sAct, 0, 0, (long long)CDK * CD, sBH, sHd, CDK, CB * CH, CH, 1.0f, 2);
    gemm(xkvh, WvT, bv, vth, CS, CDK, CD, CD, CD, CS,     // V^T epilogue
         sAct, 0, 0, (long long)CDK * CD, sBH, sHd, CDK, CB * CH, CH, 1.0f, 4);
    // scores = (q @ k^T)/8 : A=q [S,DK], B=k [S,DK] (already N x K form)
    gemm(qh, kh, nullptr, scb, CS, CS, CDK, CDK, CDK, CS,
         sBH, sHd, sBH, sHd, (long long)CH * CS * CS, (long long)CS * CS, 0,
         CB * CH, CH, 0.125f, 0);
    softmax_causal<<<CB * CH * CS, 256, 0, stream>>>(scb, CS, causal);
    // ctx = P @ V : A = f16 probs (row pitch 2S halves), B = V^T [DK,S]
    gemm((const _Float16*)scb, vth, nullptr, ctxh, CS, CDK, CS, 2 * CS, CS, CDK,
         2ll * CH * CS * CS, 2ll * CS * CS, sBH, sHd, sBH, sHd, 0,
         CB * CH, CH, 1.0f, 2);
    // cat[b,s,h*DK+:] = ctx @ Wh + bh : strided f16 epilogue folds concat
    gemm(ctxh, WhT, bh_, cath, CS, CDK, CDK, CDK, CDK, CD,
         sBH, sHd, 0, (long long)CDK * CDK, sAct, CDK, CDK, CB * CH, CH, 1.0f, 2);
    // out = cat @ Wo + bo (fp32, batch folded into M)
    gemm(cath, WoT, bo_, out, CB * CS, CD, CD, CD, CD, CD,
         0, 0, 0, 0, 0, 0, 0, 1, 1, 1.0f, 0);
  };

  for (int l = 0; l < CL; ++l) {
    auto bias4 = [&](int a, int i) {
      return (const float*)d_in[attnBase[a] + i] +
             (long long)l * ((i == 7) ? CH * CDK : (i == 9) ? CD : CH * CDK);
    };
    const float* ln1g = (const float*)d_in[44] + (long long)l * CD;
    const float* ln1b = (const float*)d_in[45] + (long long)l * CD;
    const float* ln2g = (const float*)d_in[46] + (long long)l * CD;
    const float* ln2b = (const float*)d_in[47] + (long long)l * CD;
    const float* ln3g = (const float*)d_in[48] + (long long)l * CD;
    const float* ln3b = (const float*)d_in[49] + (long long)l * CD;
    const float* b1 = (const float*)d_in[51] + (long long)l * CDF;
    const float* b2 = (const float*)d_in[53] + (long long)l * CD;

    mha(hh, hh, wqT[l][0], wkT[l][0], wvT[l][0], whT[l][0], woT[l][0],
        bias4(0, 1), bias4(0, 3), bias4(0, 5), bias4(0, 7), bias4(0, 9), 1, outa);
    mha(hh, pdh, wqT[l][1], wkT[l][1], wvT[l][1], whT[l][1], woT[l][1],
        bias4(1, 1), bias4(1, 3), bias4(1, 5), bias4(1, 7), bias4(1, 9), 1, outb);
    resid_layernorm<<<CB * CS, 256, 0, stream>>>(
        outa, outb, hbuf, 0.5f, 0.5f, 1.0f, ln1g, ln1b, r1b, r1h, CD);
    mha(r1h, ench, wqT[l][2], wkT[l][2], wvT[l][2], whT[l][2], woT[l][2],
        bias4(2, 1), bias4(2, 3), bias4(2, 5), bias4(2, 7), bias4(2, 9), 0, outa);
    mha(r1h, peh, wqT[l][3], wkT[l][3], wvT[l][3], whT[l][3], woT[l][3],
        bias4(3, 1), bias4(3, 3), bias4(3, 5), bias4(3, 7), bias4(3, 9), 0, outb);
    resid_layernorm<<<CB * CS, 256, 0, stream>>>(
        outa, outb, r1b, 0.5f, 0.5f, 1.0f, ln2g, ln2b, hbuf, hh, CD); // r2
    gemm(hh, w1T[l], b1, ffh, CB * CS, CDF, CD, CD, CD, CDF,
         0, 0, 0, 0, 0, 0, 0, 1, 1, 1.0f, 2 | 1 /*f16 + relu*/);
    gemm(ffh, w2T[l], b2, outa, CB * CS, CD, CDF, CDF, CDF, CD,
         0, 0, 0, 0, 0, 0, 0, 1, 1, 1.0f, 0);
    float* dst = (l == CL - 1) ? (float*)d_out : hbuf;
    resid_layernorm<<<CB * CS, 256, 0, stream>>>(
        outa, hbuf, nullptr, 1.0f, 1.0f, 0.0f, ln3g, ln3b, dst, hh, CD);
  }
}
